// MultiHeadAttention_10144712753741
// MI455X (gfx1250) — matrix operand order repaired and checked by reading, not compiled, not run on hardware
//
#include <hip/hip_runtime.h>
#include <hip/hip_bf16.h>
#include <math.h>

// ---------------------------------------------------------------------------
// MI455X / gfx1250 fused multi-head attention (wave32, WMMA f32_16x16x32_f16)
// ---------------------------------------------------------------------------
typedef __attribute__((ext_vector_type(16))) _Float16 v16h;
typedef __attribute__((ext_vector_type(8)))  _Float16 v8h;
typedef __attribute__((ext_vector_type(4)))  _Float16 v4h;
typedef __attribute__((ext_vector_type(8)))  float    v8f;

#define B_  4
#define S_  2048
#define D_  512
#define H_  8
#define HD_ 64
#define LOG2E_ 1.44269504088896340736f

__device__ __forceinline__ v16h mk16(v8h a, v8h b) {
  v16h r;
#pragma unroll
  for (int j = 0; j < 8; ++j) { r[j] = a[j]; r[8 + j] = b[j]; }
  return r;
}

// A-fragment (16x32 f16): lane holds row m = l16; K halves at +hi*8 and +16+hi*8
__device__ __forceinline__ v16h load_afrag(const _Float16* p) {
  v8h a0 = *(const v8h*)p;
  v8h a1 = *(const v8h*)(p + 16);
  return mk16(a0, a1);
}
// B-fragment (32x16 f16): lane holds column n; K halves at +hi*8 and +16+hi*8
__device__ __forceinline__ v16h load_bfrag(const _Float16* p) {
  v8h b0 = *(const v8h*)p;
  v8h b1 = *(const v8h*)(p + 16);
  return mk16(b0, b1);
}

__device__ __forceinline__ v8f vzero8() {
  v8f z;
#pragma unroll
  for (int i = 0; i < 8; ++i) z[i] = 0.0f;
  return z;
}

// ---------------------------------------------------------------------------
// Kernel 0: convert + transpose weights  W[K,N] f32 -> Wt[N,K] f16
// ---------------------------------------------------------------------------
__global__ void wt_kernel(const float* __restrict__ W, _Float16* __restrict__ Wt,
                          int K, int N) {
  int t = blockIdx.x * blockDim.x + threadIdx.x;
  if (t >= K * N) return;
  int k = t / N, n = t - k * N;
  Wt[(size_t)n * K + k] = (_Float16)W[t];
}

// ---------------------------------------------------------------------------
// Kernel 1/2/5: WMMA GEMM, M=8192, K=512, block = 8 waves -> 128x64 tile
//  MODE 0: X=f32 query,  W=WtQ  (N=512)  -> Qh [B,H,S,64] f16, * norm*log2e
//  MODE 1: X=f32 value,  W=WtKV (N=1024) -> Kh [B,H,S,64] f16 ;
//          Vt [B,H,64,S] f16 with intra-64 k-permutation pos=(off&15)*4+(off>>4)
//          matching the flash kernel's packed P store order.
//  MODE 2: X=f16 Xn,     W=WtO  (N=512)  -> f32 d_out [B*S, 512]
// ---------------------------------------------------------------------------
template <int MODE>
__global__ __launch_bounds__(256) void gemm_k(
    const float* __restrict__ Xf, const _Float16* __restrict__ Xh,
    const _Float16* __restrict__ WtNK, const float* __restrict__ normp,
    _Float16* __restrict__ outQ, _Float16* __restrict__ outK,
    _Float16* __restrict__ outVt, float* __restrict__ outF) {
  const int lane = threadIdx.x & 31;
  const int w    = threadIdx.x >> 5;
  const int l16  = lane & 15, hi = lane >> 4;
  const int mb   = blockIdx.x * 128 + w * 16;
  const int nb   = blockIdx.y * 64;

  v8f acc[4] = {vzero8(), vzero8(), vzero8(), vzero8()};

  for (int kc = 0; kc < D_; kc += 32) {
    v16h A;
    if (MODE == 2) {
      A = load_afrag(Xh + (size_t)(mb + l16) * D_ + kc + hi * 8);
    } else {
      const float* p = Xf + (size_t)(mb + l16) * D_ + kc + hi * 8;
#pragma unroll
      for (int j = 0; j < 8; ++j) {
        A[j]     = (_Float16)p[j];
        A[8 + j] = (_Float16)p[16 + j];
      }
    }
#pragma unroll
    for (int t = 0; t < 4; ++t) {
      v16h Bm = load_bfrag(WtNK + (size_t)(nb + t * 16 + l16) * D_ + kc + hi * 8);
      acc[t] = __builtin_amdgcn_wmma_f32_16x16x32_f16(
          false, A, false, Bm, (short)0, acc[t], false, false);
    }
  }

  if (MODE == 0) {
    const float nv = *normp * LOG2E_;  // 1/sqrt(hd) * log2(e), folded into Q
#pragma unroll
    for (int t = 0; t < 4; ++t)
#pragma unroll
      for (int g = 0; g < 8; ++g) {
        int m = mb + g + 8 * hi, n = nb + t * 16 + l16;
        int b = m >> 11, s = m & (S_ - 1);
        int h = n >> 6, d = n & 63;
        outQ[((size_t)(b * H_ + h) * S_ + s) * HD_ + d] = (_Float16)(acc[t][g] * nv);
      }
  } else if (MODE == 1) {
#pragma unroll
    for (int t = 0; t < 4; ++t)
#pragma unroll
      for (int g = 0; g < 8; ++g) {
        int m = mb + g + 8 * hi, n = nb + t * 16 + l16;
        int b = m >> 11, s = m & (S_ - 1);
        if (n < D_) {
          int h = n >> 6, d = n & 63;
          outK[((size_t)(b * H_ + h) * S_ + s) * HD_ + d] = (_Float16)acc[t][g];
        } else {
          int n2 = n - D_;
          int h = n2 >> 6, d = n2 & 63;
          // permuted-V layout: within each 64-row block of S, row 'off' goes
          // to slot (off&15)*4 + (off>>4)  (same as flash P-pack order)
          int sb  = s & ~63, off = s & 63;
          int pos = ((off & 15) << 2) | (off >> 4);
          outVt[((size_t)(b * H_ + h) * HD_ + d) * S_ + sb + pos] =
              (_Float16)acc[t][g];
        }
      }
  } else {
#pragma unroll
    for (int t = 0; t < 4; ++t)
#pragma unroll
      for (int g = 0; g < 8; ++g) {
        int m = mb + g + 8 * hi, n = nb + t * 16 + l16;
        outF[(size_t)m * D_ + n] = acc[t][g];
      }
  }
}

// ---------------------------------------------------------------------------
// Kernel 3: flash attention, causal. grid=(S/64, B*H), block=128 (4 waves).
// Wave w owns 16 query rows; 64-wide k chunks; online softmax in the WMMA
// C-register layout (row = vgpr-slot + 8*(lane>=16), col = lane%16), so only
// 16-lane shfl_xor reductions are needed. exp2-domain (log2e folded into Q).
// P is stored to LDS k-permuted (lane's 4 values packed as one b64 store);
// Vt carries the matching permutation, so PV contraction is unchanged.
// ---------------------------------------------------------------------------
__global__ __launch_bounds__(128) void flash_k(
    const _Float16* __restrict__ Qh, const _Float16* __restrict__ Kh,
    const _Float16* __restrict__ Vt, float* __restrict__ heads) {
  __shared__ __attribute__((aligned(16))) _Float16 Pbuf[4][16 * 64];

  const int lane = threadIdx.x & 31;
  const int w    = threadIdx.x >> 5;
  const int l16  = lane & 15, hi = lane >> 4;
  const int qb   = blockIdx.x * 64 + w * 16;
  const int bh   = blockIdx.y;
  const size_t skbase = (size_t)bh * S_;

  // Q fragments for d = 0..31 and 32..63 (normalizer*log2e already folded in)
  v16h QA[2];
#pragma unroll
  for (int c = 0; c < 2; ++c)
    QA[c] = load_afrag(Qh + (skbase + qb + l16) * HD_ + c * 32 + hi * 8);

  v8f o[4] = {vzero8(), vzero8(), vzero8(), vzero8()};
  float mrun[8], lrun[8];
#pragma unroll
  for (int g = 0; g < 8; ++g) { mrun[g] = -1e30f; lrun[g] = 0.0f; }

  const int nch = blockIdx.x + 1;  // 64-wide k chunks covering qb_blk+63
  for (int c = 0; c < nch; ++c) {
    const int kc = c * 64;
    // ---- scores: four 16x16 tiles over k = kc..kc+63 ----
    v8f sc[4];
#pragma unroll
    for (int t2 = 0; t2 < 4; ++t2) {
      const int kn = kc + t2 * 16 + l16;
      v8f z = vzero8();
#pragma unroll
      for (int dc = 0; dc < 2; ++dc) {
        v16h Bm = load_bfrag(Kh + (skbase + kn) * HD_ + dc * 32 + hi * 8);
        z = __builtin_amdgcn_wmma_f32_16x16x32_f16(
            false, QA[dc], false, Bm, (short)0, z, false, false);
      }
      sc[t2] = z;
    }
    if (c + 1 < nch)  // global_prefetch_b8 for next K chunk
      __builtin_prefetch(Kh + (skbase + kc + 64 + l16) * HD_, 0, 1);

    // ---- causal mask + online softmax (exp2 domain) ----
#pragma unroll
    for (int g = 0; g < 8; ++g) {
      const int q = qb + g + 8 * hi;
      float s[4];
#pragma unroll
      for (int t2 = 0; t2 < 4; ++t2)
        s[t2] = sc[t2][g] + ((kc + t2 * 16 + l16) > q ? -3e9f : 0.0f);
      float v = fmaxf(fmaxf(s[0], s[1]), fmaxf(s[2], s[3]));
#pragma unroll
      for (int off = 1; off < 16; off <<= 1) v = fmaxf(v, __shfl_xor(v, off, 16));
      float mnew  = fmaxf(mrun[g], v);
      float scale = exp2f(mrun[g] - mnew);
      float p[4], rs = 0.0f;
#pragma unroll
      for (int t2 = 0; t2 < 4; ++t2) { p[t2] = exp2f(s[t2] - mnew); rs += p[t2]; }
#pragma unroll
      for (int off = 1; off < 16; off <<= 1) rs += __shfl_xor(rs, off, 16);
      lrun[g] = lrun[g] * scale + rs;
      mrun[g] = mnew;
#pragma unroll
      for (int t = 0; t < 4; ++t) o[t][g] *= scale;
      // packed, k-permuted P store: slot = l16*4 + t2  (one b64 per row-slot)
      v4h pk;
#pragma unroll
      for (int t2 = 0; t2 < 4; ++t2) pk[t2] = (_Float16)p[t2];
      const int r = g + 8 * hi;
      *(v4h*)&Pbuf[w][r * 64 + l16 * 4] = pk;
    }
    __syncthreads();  // C-layout -> A-layout restage (uniform trip count)

    // ---- O += P * V (P: 16x64 in permuted k-slots, two 16x32 A-frags) ----
    v16h PA[2];
#pragma unroll
    for (int c2 = 0; c2 < 2; ++c2)
      PA[c2] = load_afrag(&Pbuf[w][l16 * 64 + c2 * 32 + hi * 8]);
#pragma unroll
    for (int t = 0; t < 4; ++t) {
#pragma unroll
      for (int c2 = 0; c2 < 2; ++c2) {
        v16h Bm = load_bfrag(Vt + ((size_t)bh * HD_ + t * 16 + l16) * S_ +
                             kc + c2 * 32 + hi * 8);
        o[t] = __builtin_amdgcn_wmma_f32_16x16x32_f16(
            false, PA[c2], false, Bm, (short)0, o[t], false, false);
      }
    }
    __syncthreads();
  }

  // ---- epilogue: normalize, scatter to heads[B,S,H*64] f32 ----
  const int b = bh >> 3, h = bh & 7;
#pragma unroll
  for (int g = 0; g < 8; ++g) {
    float rinv = 1.0f / lrun[g];
    const int q = qb + g + 8 * hi;
#pragma unroll
    for (int t = 0; t < 4; ++t)
      heads[((size_t)b * S_ + q) * D_ + h * HD_ + t * 16 + l16] = o[t][g] * rinv;
  }
}

// ---------------------------------------------------------------------------
// Kernel 4: cross-head normalization: one block per token (512 threads)
// ---------------------------------------------------------------------------
__global__ __launch_bounds__(512) void headnorm_k(
    const float* __restrict__ heads, const float* __restrict__ hm,
    _Float16* __restrict__ Xn) {
  __shared__ float buf[D_];
  const int t = threadIdx.x;
  const size_t base = (size_t)blockIdx.x * D_;
  float v = heads[base + t];
  buf[t] = v;
  __syncthreads();
  const int hd = t & 63;
  float s = 0.0f, s2 = 0.0f;
#pragma unroll
  for (int h = 0; h < H_; ++h) {
    float x = buf[h * HD_ + hd];
    s += x; s2 += x * x;
  }
  float mean = s * 0.125f;
  float var  = fmaxf(s2 * 0.125f - mean * mean, 0.0f);
  float xn   = (v - mean) / (sqrtf(var) + 0.01f) * hm[t >> 6];
  Xn[base + t] = (_Float16)xn;
}

// ---------------------------------------------------------------------------
// Launch. Inputs: query, value, mask(causal; applied analytically), Wq, Wkv,
// Wo, normalizer, head_mults. Workspace layout (~50 MB, 16B-aligned pieces).
// ---------------------------------------------------------------------------
extern "C" void kernel_launch(void* const* d_in, const int* in_sizes, int n_in,
                              void* d_out, int out_size, void* d_ws, size_t ws_size,
                              hipStream_t stream) {
  (void)in_sizes; (void)n_in; (void)out_size; (void)ws_size;
  const float* query = (const float*)d_in[0];
  const float* value = (const float*)d_in[1];
  const float* Wq    = (const float*)d_in[3];
  const float* Wkv   = (const float*)d_in[4];
  const float* Wo    = (const float*)d_in[5];
  const float* normp = (const float*)d_in[6];
  const float* hmul  = (const float*)d_in[7];
  float* out = (float*)d_out;

  char* ws = (char*)d_ws;
  const size_t oWtQ   = 0;
  const size_t oWtKV  = oWtQ  + (size_t)D_ * D_ * 2;         // 512 KB
  const size_t oWtO   = oWtKV + (size_t)D_ * 2 * D_ * 2;     // +1 MB
  const size_t oQ     = oWtO  + (size_t)D_ * D_ * 2;         // +512 KB
  const size_t oK     = oQ    + (size_t)B_ * S_ * D_ * 2;    // +8 MB
  const size_t oVt    = oK    + (size_t)B_ * S_ * D_ * 2;    // +8 MB
  const size_t oHeads = oVt   + (size_t)B_ * S_ * D_ * 2;    // +8 MB
  const size_t oXn    = oHeads+ (size_t)B_ * S_ * D_ * 4;    // +16 MB
  _Float16* wtq  = (_Float16*)(ws + oWtQ);
  _Float16* wtkv = (_Float16*)(ws + oWtKV);
  _Float16* wto  = (_Float16*)(ws + oWtO);
  _Float16* Qh   = (_Float16*)(ws + oQ);
  _Float16* Kh   = (_Float16*)(ws + oK);
  _Float16* Vt   = (_Float16*)(ws + oVt);
  float*    hds  = (float*)   (ws + oHeads);
  _Float16* Xn   = (_Float16*)(ws + oXn);

  // 0) weight prep
  wt_kernel<<<(D_ * D_ + 255) / 256, 256, 0, stream>>>(Wq, wtq, D_, D_);
  wt_kernel<<<(D_ * 2 * D_ + 255) / 256, 256, 0, stream>>>(Wkv, wtkv, D_, 2 * D_);
  wt_kernel<<<(D_ * D_ + 255) / 256, 256, 0, stream>>>(Wo, wto, D_, D_);
  // 1) Q projection (normalizer * log2e folded)
  gemm_k<0><<<dim3(64, 8), 256, 0, stream>>>(query, nullptr, wtq, normp,
                                             Qh, nullptr, nullptr, nullptr);
  // 2) K/V projection (V stored k-permuted per 64-block)
  gemm_k<1><<<dim3(64, 16), 256, 0, stream>>>(value, nullptr, wtkv, nullptr,
                                              nullptr, Kh, Vt, nullptr);
  // 3) causal flash attention
  flash_k<<<dim3(S_ / 64, B_ * H_), 128, 0, stream>>>(Qh, Kh, Vt, hds);
  // 4) cross-head normalization + head_mults
  headnorm_k<<<B_ * S_, D_, 0, stream>>>(hds, hmul, Xn);
  // 5) output projection -> fp32 d_out
  gemm_k<2><<<dim3(64, 8), 256, 0, stream>>>(nullptr, Xn, wto, nullptr,
                                             nullptr, nullptr, nullptr, out);
}
